// NestedGIN_66151086293455
// MI455X (gfx1250) — compile-verified, matrix operand order checked by reading, not run on hardware
//
#include <hip/hip_runtime.h>

// ---------------------------------------------------------------------------
// NestedGIN on MI455X (gfx1250, wave32).
// Projections via v_wmma_f32_16x16x32_bf16 with pre-packed bf16 operands
// (A converted once per layer, W packed fragment-major once per layer) so the
// GEMM inner loop is pure b128-load + wmma. Aggregation via f32 global
// atomics; attention + LN via wave32 shuffle/LDS.
// ---------------------------------------------------------------------------

#define NN      32768          // nodes
#define LLE     4              // set elements per node
#define NLROWS  (NN * LLE)     // 131072 rows
#define DH      128            // hidden dim
#define DKH     32             // head dim

typedef __attribute__((ext_vector_type(16))) __bf16 v16bf;
typedef __attribute__((ext_vector_type(8)))  __bf16 v8bf;
typedef __attribute__((ext_vector_type(8)))  float  v8f;

static __device__ __forceinline__ __bf16 f2bf(float f) {
  union { float f; unsigned u; } c; c.f = f;
  unsigned r = (c.u + 0x7FFFu + ((c.u >> 16) & 1u)) >> 16;  // RNE
  union { unsigned short s; __bf16 b; } o; o.s = (unsigned short)r;
  return o.b;
}

static __device__ __forceinline__ float wred32(float v) {
#pragma unroll
  for (int m = 16; m > 0; m >>= 1) v += __shfl_xor(v, m, 32);
  return v;
}

// ---------------------------------------------------------------------------
// score[b*4+l] = pred_b0 + sum_i pred_b[i]
// ---------------------------------------------------------------------------
__global__ void score_init_kernel(float* score, const float* pb0, const float* pb) {
  int i = blockIdx.x * blockDim.x + threadIdx.x;
  if (i < 256 * LLE) score[i] = pb0[0] + pb[0] + pb[1] + pb[2];
}

// ---------------------------------------------------------------------------
// m = h   (float4-vectorized copy)
// ---------------------------------------------------------------------------
__global__ void copy4_kernel(float4* __restrict__ dst, const float4* __restrict__ src, int n4) {
  int i = blockIdx.x * blockDim.x + threadIdx.x;
  if (i < n4) dst[i] = src[i];
}

// ---------------------------------------------------------------------------
// m[dst[e]] += h[src[e]]   one edge per block, one feature elem per thread.
// ---------------------------------------------------------------------------
__global__ void agg_edges_kernel(const float* __restrict__ h, const int* __restrict__ src,
                                 const int* __restrict__ dst, float* __restrict__ m) {
  int e = blockIdx.x;
  int t = threadIdx.x;
  size_t ld = (size_t)blockDim.x;
  int s = src[e], d = dst[e];
  atomicAdd(&m[(size_t)d * ld + t], h[(size_t)s * ld + t]);
}

// ---------------------------------------------------------------------------
// f32 -> bf16 row-major conversion, 8 elements per thread (b128 in, b128 out)
// ---------------------------------------------------------------------------
__global__ void cvt_bf16_kernel(const float* __restrict__ in, __bf16* __restrict__ out, int n8) {
  int i = blockIdx.x * blockDim.x + threadIdx.x;
  if (i >= n8) return;
  const float4* ip = (const float4*)in + (size_t)i * 2;
  float4 a = ip[0], b = ip[1];
  v8bf o;
  o[0] = f2bf(a.x); o[1] = f2bf(a.y); o[2] = f2bf(a.z); o[3] = f2bf(a.w);
  o[4] = f2bf(b.x); o[5] = f2bf(b.y); o[6] = f2bf(b.z); o[7] = f2bf(b.w);
  ((v8bf*)out)[i] = o;
}

// ---------------------------------------------------------------------------
// Pack W[Kdim,128] f32 into fragment-major bf16:
//   Wp[ kb ][ t ][ lane ][ e ]  (each lane's 16-elem B fragment contiguous)
// K mapping per ISA 16-bit layout: e<8 -> kb*32 + half*8 + e
//                                  e>=8 -> kb*32 + 16 + half*8 + (e-8)
// ---------------------------------------------------------------------------
__global__ void pack_w_kernel(const float* __restrict__ W, __bf16* __restrict__ Wp, int Kdim) {
  int idx = blockIdx.x * blockDim.x + threadIdx.x;
  if (idx >= Kdim * DH) return;
  int e    = idx & 15;
  int lane = (idx >> 4) & 31;
  int t    = (idx >> 9) & 7;
  int kb   = idx >> 12;
  int half = lane >> 4;
  int n    = t * 16 + (lane & 15);
  int K    = kb * 32 + ((e < 8) ? (half * 8 + e) : (16 + half * 8 + (e - 8)));
  Wp[idx] = f2bf(W[(size_t)K * DH + n]);
}

// ---------------------------------------------------------------------------
// Fused projection GEMM: out_w[M,128] = A[M,KD] @ W_w  for up to 3 weights,
// reusing the register-resident A fragments. A is bf16 row-major, W is
// packed fragment-major bf16. One wave -> 16 rows x 128 cols.
// ---------------------------------------------------------------------------
template <int KD>
__global__ __launch_bounds__(256) void wmma_proj_kernel(
    const __bf16* __restrict__ Abf,
    const __bf16* __restrict__ W0, const __bf16* __restrict__ W1,
    const __bf16* __restrict__ W2,
    float* __restrict__ O0, float* __restrict__ O1, float* __restrict__ O2) {
  constexpr int NKB = KD / 32;
  int wave = (int)((blockIdx.x * blockDim.x + threadIdx.x) >> 5);
  int lane = threadIdx.x & 31;
  int rowM = wave * 16;
  if (rowM >= NLROWS) return;
  int mrow = rowM + (lane & 15);
  int half = (lane >> 4) & 1;

  // ---- load all A fragments once (two contiguous b128 loads per k-block)
  v16bf afr[NKB];
#pragma unroll
  for (int kb = 0; kb < NKB; ++kb) {
    const __bf16* ap = Abf + (size_t)mrow * KD + kb * 32 + half * 8;
    v8bf lo = *(const v8bf*)ap;          // K = kb*32 + 8h .. +8
    v8bf hi = *(const v8bf*)(ap + 16);   // K = kb*32 + 16 + 8h .. +8
    afr[kb] = __builtin_shufflevector(lo, hi, 0, 1, 2, 3, 4, 5, 6, 7,
                                      8, 9, 10, 11, 12, 13, 14, 15);
  }

  const __bf16* Ws[3] = {W0, W1, W2};
  float*        Os[3] = {O0, O1, O2};

#pragma unroll
  for (int w = 0; w < 3; ++w) {
    if (Ws[w] == nullptr) break;
    const v16bf* wp = (const v16bf*)Ws[w];
    v8f acc[8];
#pragma unroll
    for (int t = 0; t < 8; ++t) {
#pragma unroll
      for (int i = 0; i < 8; ++i) acc[t][i] = 0.0f;
    }
#pragma unroll
    for (int kb = 0; kb < NKB; ++kb) {
#pragma unroll
      for (int t = 0; t < 8; ++t) {
        v16bf bfrag = wp[((size_t)kb * 8 + t) * 32 + lane];  // 32B contiguous
        acc[t] = __builtin_amdgcn_wmma_f32_16x16x32_bf16(
            false, afr[kb], false, bfrag, (short)0, acc[t], false, false);
      }
    }
    // ---- D layout: VGPR r -> row rowM + r + 8*half, lane&15 -> col
    float* op = Os[w];
#pragma unroll
    for (int t = 0; t < 8; ++t) {
      int col = t * 16 + (lane & 15);
#pragma unroll
      for (int r = 0; r < 8; ++r) {
        int row = rowM + r + half * 8;
        op[(size_t)row * DH + col] = acc[t][r];
      }
    }
  }
}

// ---------------------------------------------------------------------------
// Per-node set attention over L=4 elements, 4 heads of dim 32.
// One wave per node; 8 nodes per 256-thread block. Scores staged in LDS.
// ---------------------------------------------------------------------------
__global__ __launch_bounds__(256) void attn_kernel(const float* __restrict__ q,
                                                   const float* __restrict__ k,
                                                   const float* __restrict__ v,
                                                   float* __restrict__ o) {
  __shared__ float sc[8][64];   // [wave][h*16 + l*4 + m]
  int wave = threadIdx.x >> 5;
  int lane = threadIdx.x & 31;
  int n = blockIdx.x * 8 + wave;        // grid exact (32768/8 blocks)
  size_t base = (size_t)n * LLE * DH;

#pragma unroll
  for (int t = 0; t < 2; ++t) {
    int idx = lane * 2 + t;
    int h = idx >> 4, l = (idx >> 2) & 3, m = idx & 3;
    const float* qp = q + base + (size_t)l * DH + h * DKH;
    const float* kp = k + base + (size_t)m * DH + h * DKH;
    float s = 0.0f;
#pragma unroll
    for (int i = 0; i < DKH; ++i) s += qp[i] * kp[i];
    sc[wave][idx] = s * 0.17677669529663687f;  // 1/sqrt(32)
  }
  __syncthreads();

  if (lane < 16) {
    float a0 = sc[wave][lane * 4 + 0], a1 = sc[wave][lane * 4 + 1];
    float a2 = sc[wave][lane * 4 + 2], a3 = sc[wave][lane * 4 + 3];
    float mx = fmaxf(fmaxf(a0, a1), fmaxf(a2, a3));
    float e0 = __expf(a0 - mx), e1 = __expf(a1 - mx);
    float e2 = __expf(a2 - mx), e3 = __expf(a3 - mx);
    float inv = 1.0f / (e0 + e1 + e2 + e3);
    sc[wave][lane * 4 + 0] = e0 * inv; sc[wave][lane * 4 + 1] = e1 * inv;
    sc[wave][lane * 4 + 2] = e2 * inv; sc[wave][lane * 4 + 3] = e3 * inv;
  }
  __syncthreads();

#pragma unroll
  for (int l = 0; l < 4; ++l) {
#pragma unroll
    for (int c = 0; c < 4; ++c) {
      int j = c * DKH + lane;
      float acc = 0.0f;
#pragma unroll
      for (int m = 0; m < 4; ++m)
        acc += sc[wave][(c << 4) | (l << 2) | m] * v[base + (size_t)m * DH + j];
      o[base + (size_t)l * DH + j] = acc;
    }
  }
}

// ---------------------------------------------------------------------------
// Per row (n,l): x = y + bo; LN(sab); LN(ap)+relu; LN(out)+relu -> hout.
// Folds per-layer prediction head via atomicAdd into score.
// ---------------------------------------------------------------------------
__global__ __launch_bounds__(256) void postproc_kernel(
    const float* __restrict__ y, const float* __restrict__ bo,
    const float* __restrict__ sg, const float* __restrict__ sb,
    const float* __restrict__ ag, const float* __restrict__ ab,
    const float* __restrict__ og, const float* __restrict__ ob,
    const float* __restrict__ predw, const int* __restrict__ gids,
    float* __restrict__ hout, float* __restrict__ score) {
  int wave = threadIdx.x >> 5;
  int lane = threadIdx.x & 31;
  int row = blockIdx.x * 8 + wave;      // grid exact (131072/8)
  int n = row >> 2, l = row & 3;
  int c0 = lane * 4;

  float x[4];
  const float* yp = y + (size_t)row * DH;
#pragma unroll
  for (int i = 0; i < 4; ++i) x[i] = yp[c0 + i] + bo[c0 + i];

#pragma unroll
  for (int pass = 0; pass < 3; ++pass) {
    const float* g = (pass == 0) ? sg : (pass == 1) ? ag : og;
    const float* b = (pass == 0) ? sb : (pass == 1) ? ab : ob;
    float s = x[0] + x[1] + x[2] + x[3];
    float mean = wred32(s) * (1.0f / 128.0f);
    float vs = 0.0f;
#pragma unroll
    for (int i = 0; i < 4; ++i) { float d = x[i] - mean; vs += d * d; }
    float var = wred32(vs) * (1.0f / 128.0f);
    float inv = rsqrtf(var + 1e-5f);
#pragma unroll
    for (int i = 0; i < 4; ++i) {
      float t = (x[i] - mean) * inv * g[c0 + i] + b[c0 + i];
      x[i] = (pass >= 1) ? fmaxf(t, 0.0f) : t;
    }
  }

  float* hp = hout + (size_t)row * DH;
  float p = 0.0f;
#pragma unroll
  for (int i = 0; i < 4; ++i) { hp[c0 + i] = x[i]; p += x[i] * predw[c0 + i]; }
  p = wred32(p);
  if (lane == 0) atomicAdd(&score[gids[n] * LLE + l], p);
}

// ---------------------------------------------------------------------------
// Prediction head on raw features (d=64).
// ---------------------------------------------------------------------------
__global__ __launch_bounds__(256) void pred64_kernel(const float* __restrict__ h,
                                                     const float* __restrict__ pw,
                                                     const int* __restrict__ gids,
                                                     float* __restrict__ score) {
  int wave = threadIdx.x >> 5;
  int lane = threadIdx.x & 31;
  int row = blockIdx.x * 8 + wave;
  int n = row >> 2, l = row & 3;
  const float* hp = h + (size_t)row * 64;
  float p = hp[lane * 2] * pw[lane * 2] + hp[lane * 2 + 1] * pw[lane * 2 + 1];
  p = wred32(p);
  if (lane == 0) atomicAdd(&score[gids[n] * LLE + l], p);
}

// ---------------------------------------------------------------------------
extern "C" void kernel_launch(void* const* d_in, const int* in_sizes, int n_in,
                              void* d_out, int out_size, void* d_ws, size_t ws_size,
                              hipStream_t stream) {
  const float* feat   = (const float*)d_in[0];
  const int*   src    = (const int*)d_in[1];
  const int*   dst    = (const int*)d_in[2];
  const int*   gids   = (const int*)d_in[3];
  const float* wq0    = (const float*)d_in[4];
  const float* wk0    = (const float*)d_in[5];
  const float* wv0    = (const float*)d_in[6];
  const float* wo0    = (const float*)d_in[7];
  const float* bo0    = (const float*)d_in[8];
  const float* sg0    = (const float*)d_in[9];
  const float* sb0    = (const float*)d_in[10];
  const float* ag0    = (const float*)d_in[11];
  const float* ab0    = (const float*)d_in[12];
  const float* og0    = (const float*)d_in[13];
  const float* ob0    = (const float*)d_in[14];
  const float* wq     = (const float*)d_in[15];
  const float* wk     = (const float*)d_in[16];
  const float* wv     = (const float*)d_in[17];
  const float* wo     = (const float*)d_in[18];
  const float* bo     = (const float*)d_in[19];
  const float* sgs    = (const float*)d_in[20];
  const float* sbs    = (const float*)d_in[21];
  const float* ags    = (const float*)d_in[22];
  const float* abs_   = (const float*)d_in[23];
  const float* ogs    = (const float*)d_in[24];
  const float* obs    = (const float*)d_in[25];
  const float* pw0    = (const float*)d_in[26];
  const float* pb0    = (const float*)d_in[27];
  const float* pw     = (const float*)d_in[28];
  const float* pb     = (const float*)d_in[29];
  float* score = (float*)d_out;
  const int E = in_sizes[1];

  // ---- workspace layout -------------------------------------------------
  // 6 f32 buffers of 64MB, then a 32MB bf16 A buffer, then 4 packed weights.
  const size_t SZ = (size_t)NLROWS * DH * sizeof(float);
  char* ws = (char*)d_ws;
  float*  H0  = (float*)(ws + 0 * SZ);
  float*  H1  = (float*)(ws + 1 * SZ);
  float*  Mb  = (float*)(ws + 2 * SZ);   // agg result m; later reused as o
  float*  Qb  = (float*)(ws + 3 * SZ);   // q; later reused as y
  float*  Kb  = (float*)(ws + 4 * SZ);
  float*  Vb  = (float*)(ws + 5 * SZ);
  __bf16* Abf = (__bf16*)(ws + 6 * SZ);                 // bf16 activations
  const size_t WSLOT = (size_t)DH * DH;                 // 16384 bf16 = 32KB
  __bf16* Wpq = (__bf16*)(ws + 6 * SZ + SZ / 2);
  __bf16* Wpk = Wpq + WSLOT;
  __bf16* Wpv = Wpk + WSLOT;
  __bf16* Wpo = Wpv + WSLOT;

  // ---- score = bias terms; hidden_rep[0] (= feat) head ------------------
  score_init_kernel<<<4, 256, 0, stream>>>(score, pb0, pb);
  pred64_kernel<<<NLROWS / 8, 256, 0, stream>>>(feat, pw0, gids, score);

  const float* hcur = feat;
  int dd = 64;
  float* houts[3] = {H0, H1, H0};
  const int gemm_blocks = NLROWS / 16 / 8;  // wave per 16 rows, 8 waves/block

  for (int i = 0; i < 3; ++i) {
    const float *wqp, *wkp, *wvp, *wop, *bop, *sgp, *sbp, *agp, *abp, *ogp, *obp;
    if (i == 0) {
      wqp = wq0; wkp = wk0; wvp = wv0; wop = wo0; bop = bo0;
      sgp = sg0; sbp = sb0; agp = ag0; abp = ab0; ogp = og0; obp = ob0;
    } else {
      size_t r = (size_t)(i - 1);
      wqp = wq + r * DH * DH; wkp = wk + r * DH * DH; wvp = wv + r * DH * DH;
      wop = wo + r * DH * DH; bop = bo + r * DH;
      sgp = sgs + r * DH; sbp = sbs + r * DH;
      agp = ags + r * DH; abp = abs_ + r * DH;
      ogp = ogs + r * DH; obp = obs + r * DH;
    }
    const int ld = LLE * dd;              // 256 or 512
    const int n4 = NLROWS * dd / 4;
    const int n8 = NLROWS * dd / 8;

    // ---- m = h + segment_sum(h[src], dst)
    copy4_kernel<<<(n4 + 255) / 256, 256, 0, stream>>>((float4*)Mb, (const float4*)hcur, n4);
    agg_edges_kernel<<<E, ld, 0, stream>>>(hcur, src, dst, Mb);

    // ---- convert m -> bf16 once; pack the 4 weight matrices
    cvt_bf16_kernel<<<(n8 + 255) / 256, 256, 0, stream>>>(Mb, Abf, n8);
    pack_w_kernel<<<(dd * DH + 255) / 256, 256, 0, stream>>>(wqp, Wpq, dd);
    pack_w_kernel<<<(dd * DH + 255) / 256, 256, 0, stream>>>(wkp, Wpk, dd);
    pack_w_kernel<<<(dd * DH + 255) / 256, 256, 0, stream>>>(wvp, Wpv, dd);
    pack_w_kernel<<<(DH * DH + 255) / 256, 256, 0, stream>>>(wop, Wpo, DH);

    // ---- q/k/v in one fused WMMA pass (A fragments loaded once)
    if (dd == 64) {
      wmma_proj_kernel<64><<<gemm_blocks, 256, 0, stream>>>(Abf, Wpq, Wpk, Wpv,
                                                            Qb, Kb, Vb);
    } else {
      wmma_proj_kernel<128><<<gemm_blocks, 256, 0, stream>>>(Abf, Wpq, Wpk, Wpv,
                                                             Qb, Kb, Vb);
    }

    // ---- per-node set attention -> o (reuse Mb)
    attn_kernel<<<NN / 8, 256, 0, stream>>>(Qb, Kb, Vb, Mb);

    // ---- y = o @ wo (reuse Qb): convert o, single-weight WMMA pass
    cvt_bf16_kernel<<<(NLROWS * DH / 8 + 255) / 256, 256, 0, stream>>>(Mb, Abf,
                                                                       NLROWS * DH / 8);
    wmma_proj_kernel<128><<<gemm_blocks, 256, 0, stream>>>(Abf, Wpo, nullptr, nullptr,
                                                           Qb, nullptr, nullptr);

    // ---- +bias, LN x3 (+relu x2), write h_next, fold in prediction head
    postproc_kernel<<<NLROWS / 8, 256, 0, stream>>>(Qb, bop, sgp, sbp, agp, abp,
                                                    ogp, obp, pw + (size_t)i * DH,
                                                    gids, houts[i], score);
    hcur = houts[i];
    dd = DH;
  }
}
